// GlmMoeDsaModel_22986664968199
// MI455X (gfx1250) — matrix operand order repaired and checked by reading, not compiled, not run on hardware
//
#include <hip/hip_runtime.h>

#define T_TOK 1024
#define H_DIM 2048
#define E_NUM 32
#define I_DIM 512
#define SI_DIM 512

typedef __attribute__((ext_vector_type(16))) __bf16 v16bf;
typedef __attribute__((ext_vector_type(8)))  __bf16 v8bf;
typedef __attribute__((ext_vector_type(4)))  __bf16 v4bf;
typedef __attribute__((ext_vector_type(8)))  float  v8f;
typedef __attribute__((ext_vector_type(4)))  float  v4f;

// ---------------- WMMA helpers ----------------

__device__ __forceinline__ v8f wmma_bf16(v16bf a, v16bf b, v8f c) {
    // 8 args: (neg_a, A, neg_b, B, c_mod, C, reuse_a, reuse_b)
    return __builtin_amdgcn_wmma_f32_16x16x32_bf16(false, a, false, b, (short)0, c,
                                                   false, false);
}

// A-fragment (16x32 bf16, row-major source, rows = lane&15):
// lane holds K = hi*16 .. hi*16+15 (contiguous)  (hi = lane>>4)
__device__ __forceinline__ v16bf load_a_ptr(const __bf16* __restrict__ rowp, int k0) {
    union { v16bf v; v8bf h[2]; } u;
    int hoff = (threadIdx.x >> 4) * 8;
    u.h[0] = *(const v8bf*)(rowp + k0 + hoff);
    u.h[1] = *(const v8bf*)(rowp + k0 + hoff + 8);
    return u.v;
}

// B-fragment (32x16 bf16) from N x K row-major fp32 weights:
// lane holds row n = lane&15, K = hi*16 .. hi*16+15 (contiguous), cvt fp32->bf16
__device__ __forceinline__ v16bf load_b_f32(const float* __restrict__ q) {
    v4f c0 = *(const v4f*)(q);
    v4f c1 = *(const v4f*)(q + 4);
    v4f c2 = *(const v4f*)(q + 8);
    v4f c3 = *(const v4f*)(q + 12);
    v16bf r;
    r[0]  = (__bf16)c0[0]; r[1]  = (__bf16)c0[1]; r[2]  = (__bf16)c0[2]; r[3]  = (__bf16)c0[3];
    r[4]  = (__bf16)c1[0]; r[5]  = (__bf16)c1[1]; r[6]  = (__bf16)c1[2]; r[7]  = (__bf16)c1[3];
    r[8]  = (__bf16)c2[0]; r[9]  = (__bf16)c2[1]; r[10] = (__bf16)c2[2]; r[11] = (__bf16)c2[3];
    r[12] = (__bf16)c3[0]; r[13] = (__bf16)c3[1]; r[14] = (__bf16)c3[2]; r[15] = (__bf16)c3[3];
    return r;
}

// fast silu via v_exp_f32 + v_rcp_f32 (post-bf16-matmul, precision is ample)
__device__ __forceinline__ float silu_f(float x) {
    return x * __builtin_amdgcn_rcpf(1.f + __expf(-x));
}
__device__ __forceinline__ float sigmoid_f(float x) {
    return __builtin_amdgcn_rcpf(1.f + __expf(-x));
}

__device__ __forceinline__ v8f zero8() {
    v8f z = {0.f, 0.f, 0.f, 0.f, 0.f, 0.f, 0.f, 0.f};
    return z;
}

// global_prefetch_b8 (gfx1250): pull the streaming fp32 weight rows ahead of WMMA use
#define PF_DIST 128  /* floats = 512B = 4 K-iterations ahead */
__device__ __forceinline__ void prefetch_g(const float* p) {
    __builtin_prefetch((const void*)p, 0, 3);
}

// ---------------- fp32 -> bf16 activation convert ----------------

__global__ __launch_bounds__(256)
void cvt_kernel(const float* __restrict__ x, __bf16* __restrict__ xb, int n4) {
    int i = blockIdx.x * 256 + threadIdx.x;
    if (i < n4) {
        v4f v = *(const v4f*)(x + (size_t)i * 4);
        v4bf o;
        o[0] = (__bf16)v[0]; o[1] = (__bf16)v[1]; o[2] = (__bf16)v[2]; o[3] = (__bf16)v[3];
        *(v4bf*)(xb + (size_t)i * 4) = o;
    }
}

// ---------------- router: logits + sigmoid + group top-k + scatter ----------------

__global__ __launch_bounds__(256)
void router_kernel(const float* __restrict__ x, const float* __restrict__ rw,
                   const float* __restrict__ ebias, int* __restrict__ counts,
                   int* __restrict__ slot_tok, float* __restrict__ slot_w) {
    int t = blockIdx.x;
    int tid = threadIdx.x;
    int e = tid >> 3, c = tid & 7;          // 8 lanes per expert
    const float* xr = x + (size_t)t * H_DIM;
    const float* wr = rw + (size_t)e * H_DIM;
    float acc = 0.f;
    for (int h = c * 4; h < H_DIM; h += 32) {
        v4f xv = *(const v4f*)(xr + h);
        v4f wv = *(const v4f*)(wr + h);
        acc += xv[0] * wv[0] + xv[1] * wv[1] + xv[2] * wv[2] + xv[3] * wv[3];
    }
    __shared__ float red[256];
    __shared__ float logit[E_NUM];
    red[tid] = acc;
    __syncthreads();
    if (c == 0) {
        float s = 0.f;
        for (int i = 0; i < 8; i++) s += red[e * 8 + i];
        logit[e] = s;
    }
    __syncthreads();
    if (tid == 0) {
        float sc[E_NUM], sfc[E_NUM];
        for (int i = 0; i < E_NUM; i++) {
            float s = sigmoid_f(logit[i]);
            sc[i] = s;
            sfc[i] = s + ebias[i];
        }
        // group score = sum of top-2 within each group of 8
        float gs[4];
        for (int g = 0; g < 4; g++) {
            float m1 = -1e30f, m2 = -1e30f;
            for (int i = 0; i < 8; i++) {
                float v = sfc[g * 8 + i];
                if (v > m1) { m2 = m1; m1 = v; }
                else if (v > m2) { m2 = v; }
            }
            gs[g] = m1 + m2;
        }
        int g1 = 0;
        for (int g = 1; g < 4; g++) if (gs[g] > gs[g1]) g1 = g;
        int g2 = -1;
        for (int g = 0; g < 4; g++) {
            if (g == g1) continue;
            if (g2 < 0 || gs[g] > gs[g2]) g2 = g;
        }
        unsigned gmask = (1u << g1) | (1u << g2);
        int idx[4];
        float wv[4];
        float wsum = 0.f;
        bool used[E_NUM];
        for (int i = 0; i < E_NUM; i++) used[i] = false;
        for (int k = 0; k < 4; k++) {
            int best = 0;
            float bv = -1e30f;
            for (int i = 0; i < E_NUM; i++) {
                if (!((gmask >> (i >> 3)) & 1u) || used[i]) continue;
                if (sfc[i] > bv) { bv = sfc[i]; best = i; }
            }
            used[best] = true;
            idx[k] = best;
            wv[k] = sc[best];         // weights use raw sigmoid scores
            wsum += wv[k];
        }
        float scale = 2.5f / (wsum + 1e-20f);
        for (int k = 0; k < 4; k++) {
            int ee = idx[k];
            int p = atomicAdd(&counts[ee], 1);
            slot_tok[ee * T_TOK + p] = t;
            slot_w[ee * T_TOK + p] = wv[k] * scale;
        }
    }
}

__global__ void scan_kernel(const int* __restrict__ counts, int* __restrict__ basep) {
    if (threadIdx.x == 0 && blockIdx.x == 0) {
        int s = 0;
        for (int e = 0; e < E_NUM; e++) { basep[e] = s; s += counts[e]; }
    }
}

// ---------------- routed expert: gate_up GEMM + SwiGLU (64x16 tile / wave) ----------------

__global__ __launch_bounds__(32)
void gateup_kernel(const __bf16* __restrict__ xb, const float* __restrict__ gup,
                   const int* __restrict__ counts, const int* __restrict__ basep,
                   const int* __restrict__ slot_tok, const float* __restrict__ slot_w,
                   __bf16* __restrict__ hbuf) {
    int e = blockIdx.z;
    int Ne = counts[e];
    int mb = blockIdx.y * 64;
    if (mb >= Ne) return;
    int n0 = blockIdx.x * 16;
    int lane = threadIdx.x;
    int rl = lane & 15, hi = lane >> 4;
    const int* toks = slot_tok + e * T_TOK;
    // per-lane weight row pointers (B fragment: row n0+rl, contiguous K chunk at hi*16)
    const float* wgp = gup + (size_t)e * (2 * I_DIM) * H_DIM +
                       (size_t)(n0 + rl) * H_DIM + (hi << 4);
    const float* wup = wgp + (size_t)I_DIM * H_DIM;
    const __bf16* ap[4];
    for (int j = 0; j < 4; j++) {
        int m = mb + j * 16 + rl;
        int tk = toks[m < Ne ? m : 0];
        ap[j] = xb + (size_t)tk * H_DIM + hi * 8;
    }
    v8f ag[4], au[4];
    for (int j = 0; j < 4; j++) { ag[j] = zero8(); au[j] = zero8(); }
    for (int k0 = 0; k0 < H_DIM; k0 += 32) {
        if (k0 + PF_DIST < H_DIM) {
            prefetch_g(wgp + k0 + PF_DIST);
            prefetch_g(wup + k0 + PF_DIST);
        }
        v16bf bg = load_b_f32(wgp + k0);
        v16bf bu = load_b_f32(wup + k0);
#pragma unroll
        for (int j = 0; j < 4; j++) {
            v16bf a = load_a_ptr(ap[j], k0);
            ag[j] = wmma_bf16(a, bg, ag[j]);
            au[j] = wmma_bf16(a, bu, au[j]);
        }
    }
    int col = n0 + rl;
    int eb = basep[e];
    for (int j = 0; j < 4; j++) {
        int m0 = mb + j * 16 + hi * 8;
#pragma unroll
        for (int v = 0; v < 8; v++) {
            int m = m0 + v;
            if (m < Ne) {
                float g = ag[j][v], u = au[j][v];
                float hval = silu_f(g) * u * slot_w[e * T_TOK + m];
                hbuf[(size_t)(eb + m) * I_DIM + col] = (__bf16)hval;
            }
        }
    }
}

// ---------------- routed expert: down GEMM, atomic accumulate into out ----------------

__global__ __launch_bounds__(32)
void down_kernel(const __bf16* __restrict__ hbuf, const float* __restrict__ dp,
                 const int* __restrict__ counts, const int* __restrict__ basep,
                 const int* __restrict__ slot_tok, float* __restrict__ out) {
    int e = blockIdx.z;
    int Ne = counts[e];
    int mb = blockIdx.y * 64;
    if (mb >= Ne) return;
    int n0 = blockIdx.x * 16;
    int lane = threadIdx.x;
    int rl = lane & 15, hi = lane >> 4;
    const float* wdp = dp + (size_t)e * H_DIM * I_DIM + (size_t)(n0 + rl) * I_DIM + (hi << 4);
    const __bf16* A0 = hbuf + (size_t)(basep[e] + mb) * I_DIM + hi * 8;
    v8f acc[4];
    for (int j = 0; j < 4; j++) acc[j] = zero8();
    for (int k0 = 0; k0 < I_DIM; k0 += 32) {
        if (k0 + PF_DIST < I_DIM) prefetch_g(wdp + k0 + PF_DIST);
        v16bf b = load_b_f32(wdp + k0);
#pragma unroll
        for (int j = 0; j < 4; j++) {
            v16bf a = load_a_ptr(A0 + (size_t)(j * 16 + rl) * I_DIM, k0);
            acc[j] = wmma_bf16(a, b, acc[j]);
        }
    }
    int col = n0 + rl;
    const int* toks = slot_tok + e * T_TOK;
    for (int j = 0; j < 4; j++) {
        int m0 = mb + j * 16 + hi * 8;
#pragma unroll
        for (int v = 0; v < 8; v++) {
            int m = m0 + v;
            if (m < Ne) {
                int t = toks[m];
                atomicAdd(out + (size_t)t * H_DIM + col, acc[j][v]);
            }
        }
    }
}

// ---------------- shared expert ----------------

__global__ __launch_bounds__(32)
void shared_gateup_kernel(const __bf16* __restrict__ xb, const float* __restrict__ gw,
                          const float* __restrict__ uw, __bf16* __restrict__ sh) {
    int mb = blockIdx.y * 64;
    int n0 = blockIdx.x * 16;
    int lane = threadIdx.x;
    int rl = lane & 15, hi = lane >> 4;
    const __bf16* A0 = xb + (size_t)mb * H_DIM + hi * 8;
    const float* wgp = gw + (size_t)(n0 + rl) * H_DIM + (hi << 4);
    const float* wup = uw + (size_t)(n0 + rl) * H_DIM + (hi << 4);
    v8f ag[4], au[4];
    for (int j = 0; j < 4; j++) { ag[j] = zero8(); au[j] = zero8(); }
    for (int k0 = 0; k0 < H_DIM; k0 += 32) {
        if (k0 + PF_DIST < H_DIM) {
            prefetch_g(wgp + k0 + PF_DIST);
            prefetch_g(wup + k0 + PF_DIST);
        }
        v16bf bg = load_b_f32(wgp + k0);
        v16bf bu = load_b_f32(wup + k0);
#pragma unroll
        for (int j = 0; j < 4; j++) {
            v16bf a = load_a_ptr(A0 + (size_t)(j * 16 + rl) * H_DIM, k0);
            ag[j] = wmma_bf16(a, bg, ag[j]);
            au[j] = wmma_bf16(a, bu, au[j]);
        }
    }
    int col = n0 + rl;
    for (int j = 0; j < 4; j++) {
        int m0 = mb + j * 16 + hi * 8;
#pragma unroll
        for (int v = 0; v < 8; v++) {
            int m = m0 + v;
            float hval = silu_f(ag[j][v]) * au[j][v];
            sh[(size_t)m * SI_DIM + col] = (__bf16)hval;
        }
    }
}

__global__ __launch_bounds__(32)
void shared_down_kernel(const __bf16* __restrict__ sh, const float* __restrict__ dw,
                        float* __restrict__ out) {
    int mb = blockIdx.y * 64;
    int n0 = blockIdx.x * 16;
    int lane = threadIdx.x;
    int rl = lane & 15, hi = lane >> 4;
    const __bf16* A0 = sh + (size_t)mb * SI_DIM + hi * 8;
    const float* wdp = dw + (size_t)(n0 + rl) * SI_DIM + (hi << 4);
    v8f acc[4];
    for (int j = 0; j < 4; j++) acc[j] = zero8();
    for (int k0 = 0; k0 < SI_DIM; k0 += 32) {
        if (k0 + PF_DIST < SI_DIM) prefetch_g(wdp + k0 + PF_DIST);
        v16bf b = load_b_f32(wdp + k0);
#pragma unroll
        for (int j = 0; j < 4; j++) {
            v16bf a = load_a_ptr(A0 + (size_t)(j * 16 + rl) * SI_DIM, k0);
            acc[j] = wmma_bf16(a, b, acc[j]);
        }
    }
    int col = n0 + rl;
    for (int j = 0; j < 4; j++) {
        int m0 = mb + j * 16 + hi * 8;
#pragma unroll
        for (int v = 0; v < 8; v++) {
            int m = m0 + v;
            out[(size_t)m * H_DIM + col] = acc[j][v];   // initializes every out element
        }
    }
}

// ---------------- launch ----------------

extern "C" void kernel_launch(void* const* d_in, const int* in_sizes, int n_in,
                              void* d_out, int out_size, void* d_ws, size_t ws_size,
                              hipStream_t stream) {
    (void)in_sizes; (void)n_in; (void)out_size; (void)ws_size;
    const float* x   = (const float*)d_in[0];
    const float* rw  = (const float*)d_in[1];
    const float* eb  = (const float*)d_in[2];
    const float* gup = (const float*)d_in[3];
    const float* dp  = (const float*)d_in[4];
    const float* sgw = (const float*)d_in[5];
    const float* suw = (const float*)d_in[6];
    const float* sdw = (const float*)d_in[7];
    float* out = (float*)d_out;

    char* p = (char*)d_ws;
    auto carve = [&](size_t bytes) -> char* {
        char* r = p;
        p += (bytes + 255) & ~(size_t)255;
        return r;
    };
    __bf16* xb     = (__bf16*)carve((size_t)T_TOK * H_DIM * 2);
    int*    counts = (int*)carve(E_NUM * 4);
    int*    basep  = (int*)carve(E_NUM * 4);
    int*    stok   = (int*)carve((size_t)E_NUM * T_TOK * 4);
    float*  sw     = (float*)carve((size_t)E_NUM * T_TOK * 4);
    __bf16* hbuf   = (__bf16*)carve((size_t)(T_TOK * 4 + 128) * I_DIM * 2);
    __bf16* sh     = (__bf16*)carve((size_t)T_TOK * SI_DIM * 2);

    hipMemsetAsync(counts, 0, E_NUM * 4, stream);
    cvt_kernel<<<(T_TOK * H_DIM / 4 + 255) / 256, 256, 0, stream>>>(x, xb, T_TOK * H_DIM / 4);
    router_kernel<<<T_TOK, 256, 0, stream>>>(x, rw, eb, counts, stok, sw);
    scan_kernel<<<1, 32, 0, stream>>>(counts, basep);
    shared_gateup_kernel<<<dim3(SI_DIM / 16, T_TOK / 64), 32, 0, stream>>>(xb, sgw, suw, sh);
    shared_down_kernel<<<dim3(H_DIM / 16, T_TOK / 64), 32, 0, stream>>>(sh, sdw, out);
    gateup_kernel<<<dim3(I_DIM / 16, T_TOK / 64, E_NUM), 32, 0, stream>>>(
        xb, gup, counts, basep, stok, sw, hbuf);
    down_kernel<<<dim3(H_DIM / 16, T_TOK / 64, E_NUM), 32, 0, stream>>>(
        hbuf, dp, counts, basep, stok, out);
}
